// VectorQuantizer_326417515396
// MI455X (gfx1250) — matrix operand order repaired and checked by reading, not compiled, not run on hardware
//
#include <hip/hip_runtime.h>

typedef __attribute__((ext_vector_type(16))) __bf16 v16bf;
typedef __attribute__((ext_vector_type(8)))  float  v8f;

#define K_EMB    1024
#define D_EMB    256
#define HW       1024                  // 32*32
#define N_VEC    65536                 // 64*HW
#define ROWS_WG  128                   // latent vectors per workgroup
#define NUM_WG   (N_VEC / ROWS_WG)     // 512
#define KT_TILES (K_EMB / 16)          // 64 column tiles
#define D_CHUNKS (D_EMB / 32)          // 8 K-chunks per WMMA row
#define N_CHUNKS (KT_TILES * D_CHUNKS) // 512 flattened B chunks

#define LDS_A_BYTES   (ROWS_WG * D_EMB * 2)            // 64 KB
#define LDS_IDX_OFF   LDS_A_BYTES
#define LDS_RED_OFF   (LDS_A_BYTES + ROWS_WG * 4)
#define LDS_TOTAL     (LDS_RED_OFF + 4 * 4)            // 66 KB

// ---------------------------------------------------------------------------
// Repack fp32 codebook [K, D] into bf16 WMMA-B layout.
// B tile is 32(K) x 16(N), contiguous k order chosen here to match A: lanes 0-15 hold K=0..15 (2 bf16/VGPR, sequential),
// lanes 16-31 hold K=16..31; column N = lane&15. Per lane: 16 bf16 = 32 B.
// Packed index (bf16 elems): ((kt*8 + chunk)*32 + lane)*16 + j
// ---------------------------------------------------------------------------
__global__ void vq_pack_codebook(const float* __restrict__ cb,
                                 __bf16* __restrict__ cbp) {
    int i = blockIdx.x * 256 + threadIdx.x;   // one thread per (k, d)
    int k = i >> 8;
    int d = i & (D_EMB - 1);
    float v = cb[i];
    int kt = k >> 4, m = k & 15;
    int c  = d >> 5, ko = d & 31;
    int lane = m + ((ko >= 16) ? 16 : 0);
    int j = ko & 15;
    cbp[((kt * D_CHUNKS + c) * 32 + lane) * 16 + j] = (__bf16)v;
}

__global__ void vq_codebook_norms(const float* __restrict__ cb,
                                  float* __restrict__ hnorm) {
    int k = blockIdx.x * 256 + threadIdx.x;
    float s = 0.f;
    #pragma unroll 8
    for (int d = 0; d < D_EMB; ++d) {
        float v = cb[k * D_EMB + d];
        s += v * v;
    }
    hnorm[k] = 0.5f * s;   // rank by  x.e - 0.5*||e||^2  (argmax)
}

// ---------------------------------------------------------------------------
// Main kernel: 128 threads (4 waves). Each wave owns TWO 16-row A tiles
// (32 rows) -> two independent WMMA accumulator chains sharing each B tile.
// B tile loads are software-pipelined one chunk ahead.
// ---------------------------------------------------------------------------
__global__ __launch_bounds__(128) void vq_main(
        const float*  __restrict__ lat,     // [64, 256, 32, 32] fp32
        const float*  __restrict__ cb,      // [1024, 256] fp32
        const __bf16* __restrict__ cbp,     // packed bf16 codebook
        const float*  __restrict__ hnorm,   // 0.5*||e_k||^2
        float*        __restrict__ out,     // [64, 256, 32, 32] fp32
        float*        __restrict__ partials) {

    extern __shared__ __align__(32) char smem[];
    __bf16* lds_a   = (__bf16*)smem;                   // 64 KB, WMMA-A layout
    int*    lds_idx = (int*)(smem + LDS_IDX_OFF);      // 128 ints
    float*  lds_red = (float*)(smem + LDS_RED_OFF);    // 4 floats

    const int tid = threadIdx.x;
    const int wg  = blockIdx.x;
    const int n0  = wg * ROWS_WG;
    const int bi  = n0 >> 10;             // image index (HW = 1024)
    const int hw0 = n0 & (HW - 1);
    const size_t base = (size_t)bi * (D_EMB * HW) + hw0;

    // ---- Phase 1: stage latent tile to LDS in WMMA-A bf16 interleave ----
    // A tile 16(M) x 32(K), contiguous k order matching the B pack: lanes 0-15 hold K {0..15}, lanes 16-31 hold
    // K {16..31}; M = lane&15.
    #pragma unroll 4
    for (int it = 0; it < (ROWS_WG * D_EMB) / 128; ++it) {
        int i = tid + it * 128;
        int d = i >> 7;                   // 128 rows per d -> coalesced read
        int r = i & (ROWS_WG - 1);
        float v = lat[base + (size_t)d * HW + r];
        int t = r >> 4, m = r & 15;
        int c = d >> 5, ko = d & 31;
        int hi   = ko >> 4;
        int lane = m + 16 * hi;
        int j    = ko & 15;
        lds_a[((t * D_CHUNKS + c) * 32 + lane) * 16 + j] = (__bf16)v;
    }
    __syncthreads();

    // ---- Phase 2: WMMA scan over all 1024 codes; 32 rows per wave ----
    const int w    = tid >> 5;
    const int lane = tid & 31;
    const int col  = lane & 15;
    const int t0   = 2 * w;               // this wave's two A tiles

    v16bf a0[D_CHUNKS], a1[D_CHUNKS];
    #pragma unroll
    for (int c = 0; c < D_CHUNKS; ++c) {
        a0[c] = *(const v16bf*)&lds_a[(((t0 + 0) * D_CHUNKS + c) * 32 + lane) * 16];
        a1[c] = *(const v16bf*)&lds_a[(((t0 + 1) * D_CHUNKS + c) * 32 + lane) * 16];
    }

    float best0[8], best1[8];
    int   bidx0[8], bidx1[8];
    #pragma unroll
    for (int j = 0; j < 8; ++j) {
        best0[j] = -3.4e38f; bidx0[j] = 0;
        best1[j] = -3.4e38f; bidx1[j] = 0;
    }

    // Pipeline prologue: preload B chunk 0.
    v16bf bb_next = *(const v16bf*)&cbp[(size_t)lane * 16];

    for (int kt = 0; kt < KT_TILES; ++kt) {
        __builtin_prefetch(&cbp[(size_t)(((kt + 1 < KT_TILES ? kt + 1 : kt)
                                          * D_CHUNKS) * 32 + lane) * 16], 0, 1);

        v8f acc0 = (v8f){0.f, 0.f, 0.f, 0.f, 0.f, 0.f, 0.f, 0.f};
        v8f acc1 = (v8f){0.f, 0.f, 0.f, 0.f, 0.f, 0.f, 0.f, 0.f};
        #pragma unroll
        for (int c = 0; c < D_CHUNKS; ++c) {
            v16bf bb = bb_next;
            // Issue next chunk's load (crossing into next kt tile); clamp tail.
            int nc = kt * D_CHUNKS + c + 1;
            nc = (nc < N_CHUNKS) ? nc : (N_CHUNKS - 1);
            bb_next = *(const v16bf*)&cbp[((size_t)nc * 32 + lane) * 16];
            acc0 = __builtin_amdgcn_wmma_f32_16x16x32_bf16(
                       false, a0[c], false, bb, (short)0, acc0, false, false);
            acc1 = __builtin_amdgcn_wmma_f32_16x16x32_bf16(
                       false, a1[c], false, bb, (short)0, acc1, false, false);
        }

        float hn   = hnorm[kt * 16 + col];
        int   kidx = kt * 16 + col;
        #pragma unroll
        for (int j = 0; j < 8; ++j) {
            float s0 = acc0[j] - hn;      // larger s  <=>  smaller distance
            if (s0 > best0[j]) { best0[j] = s0; bidx0[j] = kidx; }
            float s1 = acc1[j] - hn;
            if (s1 > best1[j]) { best1[j] = s1; bidx1[j] = kidx; }
        }
    }

    // Cross-lane argmax over the 16 columns (stays inside each 16-lane half).
    #pragma unroll
    for (int j = 0; j < 8; ++j) {
        float bv0 = best0[j]; int bx0 = bidx0[j];
        float bv1 = best1[j]; int bx1 = bidx1[j];
        #pragma unroll
        for (int off = 1; off < 16; off <<= 1) {
            float ov0 = __shfl_xor(bv0, off, 32);
            int   ox0 = __shfl_xor(bx0, off, 32);
            if (ov0 > bv0 || (ov0 == bv0 && ox0 < bx0)) { bv0 = ov0; bx0 = ox0; }
            float ov1 = __shfl_xor(bv1, off, 32);
            int   ox1 = __shfl_xor(bx1, off, 32);
            if (ov1 > bv1 || (ov1 == bv1 && ox1 < bx1)) { bv1 = ov1; bx1 = ox1; }
        }
        if (col == 0) {
            int half = (lane >= 16) ? 8 : 0;
            lds_idx[w * 32 +      j + half] = bx0;
            lds_idx[w * 32 + 16 + j + half] = bx1;
        }
    }
    __syncthreads();

    // ---- Phase 3: fp32 gather + coalesced write-back + loss partial ----
    float lsum = 0.f;
    #pragma unroll 4
    for (int it = 0; it < (ROWS_WG * D_EMB) / 128; ++it) {
        int i = tid + it * 128;
        int d = i >> 7;
        int r = i & (ROWS_WG - 1);
        size_t addr = base + (size_t)d * HW + r;
        int   k = lds_idx[r];
        float q = cb[k * D_EMB + d];
        float x = lat[addr];
        out[addr] = q;                    // straight-through forward == q
        float df = q - x;
        lsum += df * df;
    }

    #pragma unroll
    for (int off = 1; off < 32; off <<= 1)
        lsum += __shfl_xor(lsum, off, 32);
    if (lane == 0) lds_red[w] = lsum;
    __syncthreads();
    if (tid == 0)
        partials[wg] = (lds_red[0] + lds_red[1]) + (lds_red[2] + lds_red[3]);
}

// Deterministic fixed-tree finalize: vq_loss = 1.25 * sum / numel
__global__ void vq_finalize(const float* __restrict__ partials,
                            float* __restrict__ loss_out) {
    __shared__ float red[256];
    int tid = threadIdx.x;
    float s = 0.f;
    for (int i = tid; i < NUM_WG; i += 256) s += partials[i];
    red[tid] = s;
    __syncthreads();
    for (int stride = 128; stride > 0; stride >>= 1) {
        if (tid < stride) red[tid] += red[tid + stride];
        __syncthreads();
    }
    if (tid == 0)
        loss_out[0] = 1.25f * red[0] / (float)((size_t)N_VEC * D_EMB);
}

extern "C" void kernel_launch(void* const* d_in, const int* in_sizes, int n_in,
                              void* d_out, int out_size, void* d_ws, size_t ws_size,
                              hipStream_t stream) {
    const float* lat = (const float*)d_in[0];   // latents [64,256,32,32]
    const float* cb  = (const float*)d_in[1];   // codebook [1024,256]
    float* out = (float*)d_out;                 // [64,256,32,32] ++ [loss]

    char* ws = (char*)d_ws;
    __bf16* cbp      = (__bf16*)ws;                          // 512 KB
    float*  hnorm    = (float*)(ws + 524288);                // 4 KB
    float*  partials = (float*)(ws + 524288 + 4096);         // 2 KB

    vq_pack_codebook<<<(K_EMB * D_EMB) / 256, 256, 0, stream>>>(cb, cbp);
    vq_codebook_norms<<<K_EMB / 256, 256, 0, stream>>>(cb, hnorm);
    vq_main<<<NUM_WG, 128, LDS_TOTAL, stream>>>(lat, cb, cbp, hnorm, out, partials);
    vq_finalize<<<1, 256, 0, stream>>>(partials, out + (size_t)N_VEC * D_EMB);
}
